// MultiQueryAttention_37529424233061
// MI455X (gfx1250) — matrix operand order repaired and checked by reading, not compiled, not run on hardware
//
#include <hip/hip_runtime.h>
#include <hip/hip_bf16.h>

typedef __attribute__((ext_vector_type(16))) _Float16 v16h;
typedef __attribute__((ext_vector_type(8)))  _Float16 v8h;
typedef __attribute__((ext_vector_type(8)))  float    v8f;

#define WMMA_F16(a, b, c) \
  __builtin_amdgcn_wmma_f32_16x16x32_f16(false, (a), false, (b), (short)0, (c), false, false)

static constexpr int BS = 2;
static constexpr int SEQ = 2048;
static constexpr int DIM = 1024;
static constexpr int NH = 16;
static constexpr int HD = 64;
static constexpr int MR = BS * SEQ;  // 4096 flattened rows

// ---------------- conversion kernels ----------------
__global__ void cvt_f32_f16(const float* __restrict__ in, _Float16* __restrict__ out, int n) {
  int i = blockIdx.x * blockDim.x + threadIdx.x;
  if (i < n) out[i] = (_Float16)in[i];
}

// W is [K][N] f32 row-major; write WT[N][K] f16
__global__ void cvt_transpose(const float* __restrict__ W, _Float16* __restrict__ WT,
                              int K, int N) {
  int i = blockIdx.x * blockDim.x + threadIdx.x;
  if (i < K * N) {
    int k = i / N, n = i % N;
    WT[(size_t)n * K + k] = (_Float16)W[i];
  }
}

// ---------------- generic WMMA GEMM ----------------
// C[M][N] = A[M][K] (f16) @ BT[N][K]^T (f16) + bias, epilogue by mode:
//   mode 0: f16 out, scaled by 0.125 (Q projection, folds 1/sqrt(hd))
//   mode 1: f16 out (K projection)
//   mode 2: f16 out transposed per batch: VT[(b*HD+col)*SEQ + row%SEQ]
//   mode 3: f32 out (final projection)
// Wave tile: 32(M) x 64(N). Grid must supply exactly (M/32)*(N/64) waves.
__global__ void gemm_ws(const _Float16* __restrict__ A, const _Float16* __restrict__ BT,
                        const float* __restrict__ bias, void* __restrict__ outp,
                        int M, int N, int K, int mode) {
  int lane = threadIdx.x & 31;
  int hi   = lane >> 4;
  int l16  = lane & 15;
  int wid  = (blockIdx.x * blockDim.x + threadIdx.x) >> 5;
  int ntiles = N >> 6;
  int mt = wid / ntiles;
  int nt = wid % ntiles;
  if (mt * 32 >= M) return;

  v8f c[2][4] = {};

  for (int kk = 0; kk < K; kk += 32) {
    v16h a[2];
#pragma unroll
    for (int s = 0; s < 2; ++s) {
      const _Float16* ap = A + (size_t)(mt * 32 + s * 16 + l16) * K + kk + (hi ? 16 : 0);
      v8h lo = *(const v8h*)ap;
      v8h h8 = *(const v8h*)(ap + 8);
#pragma unroll
      for (int i = 0; i < 8; ++i) { a[s][i] = lo[i]; a[s][8 + i] = h8[i]; }
    }
    v16h b[4];
#pragma unroll
    for (int n = 0; n < 4; ++n)
      b[n] = *(const v16h*)(BT + (size_t)(nt * 64 + n * 16 + l16) * K + kk + (hi ? 16 : 0));
#pragma unroll
    for (int s = 0; s < 2; ++s)
#pragma unroll
      for (int n = 0; n < 4; ++n)
        c[s][n] = WMMA_F16(a[s], b[n], c[s][n]);
  }

#pragma unroll
  for (int s = 0; s < 2; ++s)
#pragma unroll
    for (int n = 0; n < 4; ++n) {
      int col = nt * 64 + n * 16 + l16;
      float bb = bias[col];
      int row0 = mt * 32 + s * 16 + hi * 8;
#pragma unroll
      for (int j = 0; j < 8; ++j) {
        int row = row0 + j;
        float v = c[s][n][j] + bb;
        if (mode == 0) {
          ((_Float16*)outp)[(size_t)row * N + col] = (_Float16)(v * 0.125f);
        } else if (mode == 1) {
          ((_Float16*)outp)[(size_t)row * N + col] = (_Float16)v;
        } else if (mode == 2) {
          int bidx = row >> 11;            // row / SEQ
          int sr   = row & (SEQ - 1);
          ((_Float16*)outp)[((size_t)(bidx * HD + col)) * SEQ + sr] = (_Float16)v;
        } else {
          ((float*)outp)[(size_t)row * N + col] = v;
        }
      }
    }
}

// ---------------- flash attention (MQA) ----------------
// One wave per (b, h, 16-row q tile). S^T = K*Q^T so the P fragment maps
// directly to the PV WMMA A-operand without cross-lane movement.
__global__ void mqa_attn(const _Float16* __restrict__ Qh, const _Float16* __restrict__ Kh,
                         const _Float16* __restrict__ VT, _Float16* __restrict__ Ah) {
  int lane = threadIdx.x & 31;
  int hi   = lane >> 4;
  int l16  = lane & 15;
  int wid  = (blockIdx.x * blockDim.x + threadIdx.x) >> 5;
  int qt = wid & 127;
  int h  = (wid >> 7) & (NH - 1);
  int b  = wid >> 11;
  int qbase = qt * 16;

  // Q fragments (B operand of S^T wmma): lane holds col q = qbase+l16,
  // element i = hd-dim = ks*32 + hi*16 + i  -> contiguous 16 halves
  v16h qf[2];
#pragma unroll
  for (int ks = 0; ks < 2; ++ks)
    qf[ks] = *(const v16h*)(Qh + (size_t)(b * SEQ + qbase + l16) * DIM + h * HD + ks * 32 + hi * 16);

  v8f o[4] = {};
  float m = -__builtin_inff();
  float l = 0.0f;
  int q = qbase + l16;  // this lane's query row (softmax state is per-q)

  int kbmax = (qbase + 15) >> 5;  // inclusive 32-wide key block index
  for (int kb = 0; kb <= kbmax; ++kb) {
    int kstart = kb * 32;
    v8f sfrag[2] = {};

    // S^T: A = K (rows = keys, kdim = hd), accumulate over hd in 2 steps
#pragma unroll
    for (int ksub = 0; ksub < 2; ++ksub) {
      int krow = kstart + ksub * 16 + l16;
#pragma unroll
      for (int ks = 0; ks < 2; ++ks) {
        const _Float16* kp = Kh + (size_t)(b * SEQ + krow) * HD + ks * 32 + (hi ? 16 : 0);
        v8h lo = *(const v8h*)kp;
        v8h h8 = *(const v8h*)(kp + 8);
        v16h af;
#pragma unroll
        for (int i = 0; i < 8; ++i) { af[i] = lo[i]; af[8 + i] = h8[i]; }
        sfrag[ksub] = WMMA_F16(af, qf[ks], sfrag[ksub]);
      }
    }

    // causal mask: element (key = kstart + ksub*16 + hi*8 + j, col q)
    if (kstart + 31 > qbase) {
#pragma unroll
      for (int ksub = 0; ksub < 2; ++ksub)
#pragma unroll
        for (int j = 0; j < 8; ++j) {
          int key = kstart + ksub * 16 + hi * 8 + j;
          if (key > q) sfrag[ksub][j] = -__builtin_inff();
        }
    }

    // online softmax (state per q = l16, replicated across lane^16)
    float bm = -__builtin_inff();
#pragma unroll
    for (int ksub = 0; ksub < 2; ++ksub)
#pragma unroll
      for (int j = 0; j < 8; ++j) bm = fmaxf(bm, sfrag[ksub][j]);
    bm = fmaxf(bm, __shfl_xor(bm, 16));
    float mnew  = fmaxf(m, bm);
    float alpha = __expf(m - mnew);  // first iter: exp(-inf)=0, o is 0 anyway

    float rowsum = 0.0f;
    v16h pa;  // P as A-operand: elements 0-7 = frag0, 8-15 = frag1 (exact layout match)
#pragma unroll
    for (int ksub = 0; ksub < 2; ++ksub)
#pragma unroll
      for (int j = 0; j < 8; ++j) {
        float p = __expf(sfrag[ksub][j] - mnew);
        rowsum += p;
        pa[ksub * 8 + j] = (_Float16)p;
      }
    rowsum += __shfl_xor(rowsum, 16);
    l = l * alpha + rowsum;
    m = mnew;

    // rescale accumulator rows (row q' = hi*8 + j lives in lane l16=q')
    float aj[8];
#pragma unroll
    for (int j = 0; j < 8; ++j) aj[j] = __shfl(alpha, hi * 8 + j);
#pragma unroll
    for (int n = 0; n < 4; ++n) {
#pragma unroll
      for (int j = 0; j < 8; ++j) o[n][j] *= aj[j];
      const _Float16* vp = VT + (size_t)(b * HD + n * 16 + l16) * SEQ + kstart + hi * 8;
      v8h vlo = *(const v8h*)vp;
      v8h vhi = *(const v8h*)(vp + 16);
      v16h vb;
#pragma unroll
      for (int i = 0; i < 8; ++i) { vb[i] = vlo[i]; vb[8 + i] = vhi[i]; }
      o[n] = WMMA_F16(pa, vb, o[n]);
    }
  }

  // finalize: divide row q' by its softmax sum and store to Ah[b][q'][h*64+col]
#pragma unroll
  for (int j = 0; j < 8; ++j) {
    float lj  = __shfl(l, hi * 8 + j);
    float inv = 1.0f / lj;
    int row = qbase + hi * 8 + j;
#pragma unroll
    for (int n = 0; n < 4; ++n)
      Ah[(size_t)(b * SEQ + row) * DIM + h * HD + n * 16 + l16] = (_Float16)(o[n][j] * inv);
  }
}

// ---------------- launch ----------------
extern "C" void kernel_launch(void* const* d_in, const int* in_sizes, int n_in,
                              void* d_out, int out_size, void* d_ws, size_t ws_size,
                              hipStream_t stream) {
  const float* X  = (const float*)d_in[0];
  const float* Wq = (const float*)d_in[1];
  const float* bq = (const float*)d_in[2];
  const float* Wk = (const float*)d_in[3];
  const float* bk = (const float*)d_in[4];
  const float* Wv = (const float*)d_in[5];
  const float* bv = (const float*)d_in[6];
  const float* Wo = (const float*)d_in[7];
  const float* bo = (const float*)d_in[8];

  _Float16* ws  = (_Float16*)d_ws;
  _Float16* Xh  = ws;                                  // 4096*1024
  _Float16* Qh  = Xh  + (size_t)MR * DIM;              // 4096*1024
  _Float16* Ah  = Qh  + (size_t)MR * DIM;              // 4096*1024
  _Float16* WqT = Ah  + (size_t)MR * DIM;              // 1024*1024
  _Float16* WoT = WqT + (size_t)DIM * DIM;             // 1024*1024
  _Float16* WkT = WoT + (size_t)DIM * DIM;             // 64*1024
  _Float16* WvT = WkT + (size_t)HD * DIM;              // 64*1024
  _Float16* Kh  = WvT + (size_t)HD * DIM;              // 4096*64
  _Float16* VT  = Kh  + (size_t)MR * HD;               // 2*64*2048

  // stage 1: conversions
  cvt_f32_f16<<<(MR * DIM + 255) / 256, 256, 0, stream>>>(X, Xh, MR * DIM);
  cvt_transpose<<<(DIM * DIM + 255) / 256, 256, 0, stream>>>(Wq, WqT, DIM, DIM);
  cvt_transpose<<<(DIM * HD + 255) / 256, 256, 0, stream>>>(Wk, WkT, DIM, HD);
  cvt_transpose<<<(DIM * HD + 255) / 256, 256, 0, stream>>>(Wv, WvT, DIM, HD);
  cvt_transpose<<<(DIM * DIM + 255) / 256, 256, 0, stream>>>(Wo, WoT, DIM, DIM);

  // stage 2: projections (waves = (M/32)*(N/64); 8 waves per 256-thread block)
  gemm_ws<<<(MR / 32) * (DIM / 64) / 8, 256, 0, stream>>>(Xh, WqT, bq, Qh, MR, DIM, DIM, 0);
  gemm_ws<<<(MR / 32) * (HD / 64) / 8, 256, 0, stream>>>(Xh, WkT, bk, Kh, MR, HD, DIM, 1);
  gemm_ws<<<(MR / 32) * (HD / 64) / 8, 256, 0, stream>>>(Xh, WvT, bv, VT, MR, HD, DIM, 2);

  // stage 3: attention — 2*16*128 = 4096 waves
  mqa_attn<<<BS * NH * (SEQ / 16) / 8, 256, 0, stream>>>(Qh, Kh, VT, Ah);

  // stage 4: output projection, f32 out
  gemm_ws<<<(MR / 32) * (DIM / 64) / 8, 256, 0, stream>>>(Ah, WoT, bo, d_out, MR, DIM, DIM, 3);
}
